// Texformer_377957122756
// MI455X (gfx1250) — matrix operand order repaired and checked by reading, not compiled, not run on hardware
//
#include <hip/hip_runtime.h>

// ---------------------------------------------------------------------------
// Texformer for MI455X (gfx1250). All matmul-shaped work (3x3 convs as
// implicit GEMM, 1x1 convs, linear-attention GEMMs) runs on
// v_wmma_f32_16x16x32_bf16 (bf16 in / f32 acc). fp32->bf16 conversion happens
// only in LDS staging / pack kernels. Bias + LeakyReLU + residual + nearest-2x
// upsample are fused into conv epilogues/staging (memory-bound workload).
// ---------------------------------------------------------------------------

typedef __bf16 bf16_t;
typedef __attribute__((ext_vector_type(16))) __bf16 v16bf;
typedef __attribute__((ext_vector_type(8)))  __bf16 v8bf;
typedef __attribute__((ext_vector_type(8)))  float  v8f;

__device__ inline v8f vzero() {
  v8f v;
#pragma unroll
  for (int i = 0; i < 8; i++) v[i] = 0.f;
  return v;
}

__device__ inline v8f wmma_bf16(v16bf a, v16bf b, v8f c) {
  // D = A(16x32) * B(32x16) + C(16x16), f32 accumulate
  return __builtin_amdgcn_wmma_f32_16x16x32_bf16(false, a, false, b, (short)0, c,
                                                 false, false);
}

__device__ inline float lrelu_f(float v) { return v >= 0.f ? v : 0.2f * v; }

// A-matrix (16x32 bf16) element->K mapping used here: contiguous k = h + 16*(lane/16), same as B
__device__ __host__ inline int ka_map(int h, int g) {
  return h + 16 * g;
}

// ---------------------------------------------------------------------------
// Weight pack: OIHW fp32 -> fragment-linear bf16.
// K ordering: k = tap*Cin + cin, tap = ky*3+kx (KS=9) or 0 (KS=1).
// wp[((kc*MT+mt)*32 + lane)*16 + h] = W[K = kc*32 + Ka(h, lane>>4)]
//                                      [cout = mt*16 + (lane&15)]
// ---------------------------------------------------------------------------
__global__ void k_pack(const float* __restrict__ w, bf16_t* __restrict__ wp,
                       int Cout, int Cin, int KS, int KC, int MT) {
  int idx = blockIdx.x * 256 + threadIdx.x;
  int tot = KC * MT * 512;
  if (idx >= tot) return;
  int h = idx & 15;
  int lane = (idx >> 4) & 31;
  int mt = (idx >> 9) % MT;
  int kc = (idx >> 9) / MT;
  int g = lane >> 4;
  int K = kc * 32 + ka_map(h, g);
  int cout = mt * 16 + (lane & 15);
  int Ktot = Cin * KS;
  float v = 0.f;
  if (K < Ktot && cout < Cout) {
    int cin = K % Cin;
    int tap = K / Cin;
    v = w[(cout * Cin + cin) * KS + tap];
  }
  wp[idx] = (bf16_t)v;
}

// ---------------------------------------------------------------------------
// 3x3 conv, pad=1, implicit GEMM. Block = 128 threads = 4 waves.
// Wave computes Cout(<=64) x 16 pixels. Input row-window staged to LDS bf16
// in [ky][x][cin] order -> B-fragment is one 32B contiguous ds load / lane.
// UP: nearest-2x upsample fused into staging index math.
// ---------------------------------------------------------------------------
template <int CIN, int STRIDE, bool UP>
__global__ __launch_bounds__(128) void k_conv3x3(
    const float* __restrict__ in, const bf16_t* __restrict__ wp,
    const float* __restrict__ bias, const float* __restrict__ res,
    float* __restrict__ out, int hin, int win, int hout, int wout, int Cout,
    int MT, int relu) {
  constexpr int LW = 64 * STRIDE + 2;
  __shared__ __align__(32) bf16_t lds[3 * LW * CIN];
  const int tid = threadIdx.x;
  const int b = blockIdx.z, y = blockIdx.y, x0 = blockIdx.x * 64;
  const int hie = UP ? hin * 2 : hin;
  const int wie = UP ? win * 2 : win;
  const int tot = 3 * LW * CIN;
  for (int i = tid; i < tot; i += 128) {
    int cin = i % CIN;
    int rest = i / CIN;
    int ixl = rest % LW;
    int ky = rest / LW;
    int iy = y * STRIDE + ky - 1;
    int ix = x0 * STRIDE + ixl - 1;
    float v = 0.f;
    if (iy >= 0 && iy < hie && ix >= 0 && ix < wie) {
      int sy = UP ? (iy >> 1) : iy;
      int sx = UP ? (ix >> 1) : ix;
      v = in[(((size_t)b * CIN + cin) * hin + sy) * win + sx];
    }
    lds[i] = (bf16_t)v;
  }
  __syncthreads();

  const int lane = tid & 31, wvid = tid >> 5;
  const int ln = lane & 15, g = lane >> 4;
  const int xl = wvid * 16 + ln;  // this lane's output pixel (N slot)
  constexpr int KC = (CIN * 9 + 31) / 32;
  v8f acc[4];
  for (int i = 0; i < 4; i++) acc[i] = vzero();
  const int ntiles = (Cout + 15) >> 4;

  for (int kc = 0; kc < KC; kc++) {
    v16bf bf;
    if (CIN == 64) {
      int kb2 = kc * 32 + 16 * g;        // 16 consecutive K for this half-wave
      int tap = kb2 >> 6;                // same tap for all 16
      int cin0 = kb2 & 63;
      int ky = tap / 3, kx = tap - ky * 3;
      int xw = xl * STRIDE + kx;
      bf = *(const v16bf*)&lds[(ky * LW + xw) * 64 + cin0];
    } else {
#pragma unroll
      for (int h = 0; h < 16; h++) {
        int K = kc * 32 + h + 16 * g;
        bf16_t v = (bf16_t)0.f;
        if (K < CIN * 9) {
          int cin = K % CIN;
          int tap = K / CIN;
          int ky = tap / 3, kx = tap - ky * 3;
          v = lds[((ky * LW + xl * STRIDE + kx) * CIN) + cin];
        }
        bf[h] = v;
      }
    }
    for (int mt = 0; mt < ntiles; mt++) {
      v16bf a = *(const v16bf*)&wp[(((size_t)kc * MT + mt) * 32 + lane) * 16];
      acc[mt] = wmma_bf16(a, bf, acc[mt]);
    }
  }

  const int x = x0 + xl;
  for (int mt = 0; mt < ntiles; mt++) {
#pragma unroll
    for (int r = 0; r < 8; r++) {
      int cout = mt * 16 + r + 8 * g;
      if (cout >= Cout) continue;
      float v = acc[mt][r];
      if (bias) v += bias[cout];
      if (relu) v = lrelu_f(v);
      size_t o = (((size_t)b * Cout + cout) * hout + y) * wout + x;
      if (res) v += res[o];
      out[o] = v;
    }
  }
}

// ---------------------------------------------------------------------------
// Pointwise (1x1) conv as GEMM over pixels. Same wave tiling as above.
// ---------------------------------------------------------------------------
template <int CIN>
__global__ __launch_bounds__(128) void k_pw(
    const float* __restrict__ in, const bf16_t* __restrict__ wp,
    const float* __restrict__ bias, const float* __restrict__ res,
    float* __restrict__ out, int HW, int Cout, int MT, int relu) {
  __shared__ __align__(32) bf16_t lds[64 * CIN];
  const int tid = threadIdx.x;
  const int b = blockIdx.z;
  const int p0 = blockIdx.x * 64;
  for (int i = tid; i < 64 * CIN; i += 128) {
    int cin = i % CIN;
    int xl = i / CIN;
    lds[i] = (bf16_t)in[((size_t)b * CIN + cin) * HW + p0 + xl];
  }
  __syncthreads();
  const int lane = tid & 31, wvid = tid >> 5;
  const int ln = lane & 15, g = lane >> 4;
  const int xl = wvid * 16 + ln;
  v8f acc[8];
  for (int i = 0; i < 8; i++) acc[i] = vzero();
  const int ntiles = (Cout + 15) >> 4;
  constexpr int KC = CIN / 32;
  for (int kc = 0; kc < KC; kc++) {
    int cin0 = kc * 32 + 16 * g;
    v16bf bf = *(const v16bf*)&lds[xl * CIN + cin0];
    for (int mt = 0; mt < ntiles; mt++) {
      v16bf a = *(const v16bf*)&wp[(((size_t)kc * MT + mt) * 32 + lane) * 16];
      acc[mt] = wmma_bf16(a, bf, acc[mt]);
    }
  }
  const int p = p0 + xl;
  for (int mt = 0; mt < ntiles; mt++) {
#pragma unroll
    for (int r = 0; r < 8; r++) {
      int cout = mt * 16 + r + 8 * g;
      if (cout >= Cout) continue;
      float v = acc[mt][r];
      if (bias) v += bias[cout];
      if (relu) v = lrelu_f(v);
      size_t o = ((size_t)b * Cout + cout) * HW + p;
      if (res) v += res[o];
      out[o] = v;
    }
  }
}

// ---------------------------------------------------------------------------
// Batched NT GEMM: C[M,N] = scale * A[M,Kd] * B[N,Kd]^T  (bf16 in, f32 acc)
// If Cf != null: store fp32 row-major [bn][M][N].
// Else: store bf16 transposed [bn][N][M]   (used to produce tmp^T directly).
// ---------------------------------------------------------------------------
__global__ __launch_bounds__(128) void k_gemm_nt(
    const bf16_t* __restrict__ A, const bf16_t* __restrict__ Bm,
    float* __restrict__ Cf, bf16_t* __restrict__ Ct, int M, int N, int Kd,
    float scale) {
  const int bn = blockIdx.z;
  const bf16_t* Ab = A + (size_t)bn * M * Kd;
  const bf16_t* Bb = Bm + (size_t)bn * N * Kd;
  const int lane = threadIdx.x & 31, wvid = threadIdx.x >> 5;
  const int ln = lane & 15, g = lane >> 4;
  const int m0 = blockIdx.y * 64 + wvid * 16;
  const int n0 = blockIdx.x * 64;
  v8f acc[4];
  for (int i = 0; i < 4; i++) acc[i] = vzero();
  for (int kb = 0; kb < Kd; kb += 32) {
    const bf16_t* pa = Ab + (size_t)(m0 + ln) * Kd + kb + 16 * g;
    v8bf a0 = *(const v8bf*)pa;
    v8bf a1 = *(const v8bf*)(pa + 8);
    v16bf a;
#pragma unroll
    for (int h = 0; h < 8; h++) {
      a[h] = a0[h];
      a[h + 8] = a1[h];
    }
#pragma unroll
    for (int nt = 0; nt < 4; nt++) {
      v16bf bf = *(const v16bf*)(Bb + (size_t)(n0 + nt * 16 + ln) * Kd + kb + 16 * g);
      acc[nt] = wmma_bf16(a, bf, acc[nt]);
    }
  }
#pragma unroll
  for (int nt = 0; nt < 4; nt++) {
#pragma unroll
    for (int r = 0; r < 8; r++) {
      int m = m0 + r + 8 * g;
      int n = n0 + nt * 16 + ln;
      float v = acc[nt][r] * scale;
      if (Cf)
        Cf[(size_t)bn * M * N + (size_t)m * N + n] = v;
      else
        Ct[(size_t)bn * M * N + (size_t)n * M + m] = (bf16_t)v;
    }
  }
}

// ---------------------------------------------------------------------------
// space_to_depth gather: src [B,64,h,w] fp32 -> bf16 [bn=16][c'=512][S]
// (or transposed [bn][S][512] for Q). c' = d*64 + py*8 + px, ch = n*8+d.
// ---------------------------------------------------------------------------
__global__ void k_gather(const float* __restrict__ src, bf16_t* __restrict__ dst,
                         int hs, int ws, int transposed) {
  size_t idx = (size_t)blockIdx.x * 256 + threadIdx.x;
  const int WB = ws >> 3;
  const int S = (hs >> 3) * WB;
  size_t tot = (size_t)16 * 512 * S;
  if (idx >= tot) return;
  int s = (int)(idx % S);
  size_t r = idx / S;
  int c = (int)(r % 512);
  int bn = (int)(r / 512);
  int b = bn >> 3, n = bn & 7;
  int d = c >> 6, py = (c >> 3) & 7, px = c & 7;
  int hb = s / WB, wb = s % WB;
  float v = src[(((size_t)b * 64 + n * 8 + d) * hs + hb * 8 + py) * ws + wb * 8 + px];
  if (transposed)
    dst[((size_t)bn * S + s) * 512 + c] = (bf16_t)v;
  else
    dst[idx] = (bf16_t)v;
}

// depth_to_space scatter (pixel_shuffle): g2 fp32 [bn][512][S] -> [B,64,h,w]
__global__ void k_scatter(const float* __restrict__ g2, float* __restrict__ dst,
                          int hs, int ws) {
  size_t idx = (size_t)blockIdx.x * 256 + threadIdx.x;
  const int WB = ws >> 3;
  const int S = (hs >> 3) * WB;
  size_t tot = (size_t)16 * 512 * S;
  if (idx >= tot) return;
  int s = (int)(idx % S);
  size_t r = idx / S;
  int c = (int)(r % 512);
  int bn = (int)(r / 512);
  int b = bn >> 3, n = bn & 7;
  int d = c >> 6, py = (c >> 3) & 7, px = c & 7;
  int hb = s / WB, wb = s % WB;
  dst[(((size_t)b * 64 + n * 8 + d) * hs + hb * 8 + py) * ws + wb * 8 + px] = g2[idx];
}

// DETR sine positional encoding (npf=32, normalize=True), fused add
__global__ void k_addpe(const float* __restrict__ in, float* __restrict__ out,
                        int hs, int ws) {
  size_t idx = (size_t)blockIdx.x * 256 + threadIdx.x;
  size_t tot = (size_t)2 * 64 * hs * ws;
  if (idx >= tot) return;
  int x = (int)(idx % ws);
  int y = (int)((idx / ws) % hs);
  int c = (int)((idx / ((size_t)ws * hs)) % 64);
  float coord, dimn;
  int cc;
  if (c < 32) { coord = (float)(y + 1); dimn = (float)hs; cc = c; }
  else        { coord = (float)(x + 1); dimn = (float)ws; cc = c - 32; }
  float e = coord / (dimn + 1e-6f) * 6.28318530717958647692f;
  float t = powf(10000.f, (float)(cc & ~1) * (1.f / 32.f));
  float a = e / t;
  float pe = (cc & 1) ? cosf(a) : sinf(a);
  out[idx] = in[idx] + pe;
}

// GroupNorm(1 group) deterministic two-pass reduction
__global__ void k_gn_part(const float* __restrict__ x, float* __restrict__ part,
                          int n) {
  int b = blockIdx.y;
  float s = 0.f, s2 = 0.f;
  for (size_t i = (size_t)blockIdx.x * 256 + threadIdx.x; i < (size_t)n;
       i += (size_t)gridDim.x * 256) {
    float v = x[(size_t)b * n + i];
    s += v;
    s2 += v * v;
  }
  __shared__ float sh[256], sh2[256];
  sh[threadIdx.x] = s;
  sh2[threadIdx.x] = s2;
  __syncthreads();
  for (int st = 128; st > 0; st >>= 1) {
    if ((int)threadIdx.x < st) {
      sh[threadIdx.x] += sh[threadIdx.x + st];
      sh2[threadIdx.x] += sh2[threadIdx.x + st];
    }
    __syncthreads();
  }
  if (threadIdx.x == 0) {
    part[(b * 128 + blockIdx.x) * 2 + 0] = sh[0];
    part[(b * 128 + blockIdx.x) * 2 + 1] = sh2[0];
  }
}

__global__ void k_gn_fin(const float* __restrict__ part, float* __restrict__ red) {
  if (threadIdx.x == 0) {
    for (int b = 0; b < 2; b++) {
      float s = 0.f, s2 = 0.f;
      for (int i = 0; i < 128; i++) {
        s += part[(b * 128 + i) * 2 + 0];
        s2 += part[(b * 128 + i) * 2 + 1];
      }
      red[b * 2 + 0] = s;
      red[b * 2 + 1] = s2;
    }
  }
}

__global__ void k_gn_apply(const float* __restrict__ x, const float* __restrict__ red,
                           const float* __restrict__ gw, const float* __restrict__ gb,
                           float* __restrict__ out, int n, int S) {
  size_t idx = (size_t)blockIdx.x * 256 + threadIdx.x;
  size_t tot = (size_t)2 * n;
  if (idx >= tot) return;
  int b = (int)(idx / n);
  int rem = (int)(idx % n);
  int c = rem / S;
  float inv_n = 1.f / (float)n;
  float mu = red[b * 2] * inv_n;
  float var = red[b * 2 + 1] * inv_n - mu * mu;
  float xn = (x[idx] - mu) * rsqrtf(var + 1e-5f);
  out[idx] = xn * gw[c] + gb[c];
}

// ---------------------------------------------------------------------------
// Host orchestration
// ---------------------------------------------------------------------------
extern "C" void kernel_launch(void* const* d_in, const int* in_sizes, int n_in,
                              void* d_out, int out_size, void* d_ws, size_t ws_size,
                              hipStream_t stream) {
  (void)in_sizes; (void)n_in; (void)out_size; (void)ws_size;

  // ---- parse inputs (JAX pytree flatten: sorted dict keys at every level) ----
  // top: k, params(conv0..5, conv6, conv_out, trans, unet_k, unet_q), q, v
  struct CpT { const float* b; const float* w; };
  int ip = 0;
  auto nxt = [&]() -> const float* { return (const float*)d_in[ip++]; };
  const float* in_k = nxt();
  CpT convd[6][2];
  for (int l = 0; l < 6; l++)
    for (int j = 0; j < 2; j++) { convd[l][j].b = nxt(); convd[l][j].w = nxt(); }
  CpT conv6p;   conv6p.b = nxt();   conv6p.w = nxt();
  CpT convOutp; convOutp.b = nxt(); convOutp.w = nxt();
  struct TransT {
    const float *fc, *wk, *wq, *wv, *gn_b, *gn_w;
    CpT mlp0, mlp1;
  } tr[3];
  for (int i = 0; i < 3; i++) {  // attn keys sorted: fc,wk,wq,wv; then gn_b,gn_w; mlp
    tr[i].fc = nxt(); tr[i].wk = nxt(); tr[i].wq = nxt(); tr[i].wv = nxt();
    tr[i].gn_b = nxt(); tr[i].gn_w = nxt();
    tr[i].mlp0.b = nxt(); tr[i].mlp0.w = nxt();
    tr[i].mlp1.b = nxt(); tr[i].mlp1.w = nxt();
  }
  struct UnetT { CpT c[6][2]; CpT cin; } uk, uq;
  auto parseU = [&](UnetT& u) {  // sorted: c1..c6 then cin
    for (int l = 0; l < 6; l++)
      for (int j = 0; j < 2; j++) { u.c[l][j].b = nxt(); u.c[l][j].w = nxt(); }
    u.cin.b = nxt(); u.cin.w = nxt();
  };
  parseU(uk);
  parseU(uq);
  const float* in_q = nxt();
  const float* in_v = nxt();

  // ---- workspace bump allocator (deterministic; ~440MB total) ----
  size_t wsOff = 0;
  auto wsAlloc = [&](size_t bytes) -> void* {
    void* p = (char*)d_ws + wsOff;
    wsOff = (wsOff + bytes + 255) & ~(size_t)255;
    return p;
  };
  const size_t F256 = (size_t)2 * 64 * 256 * 256;  // 8.39M floats
  const size_t F128 = F256 / 4, F64 = F256 / 16;
  float *P3[3], *P4[3], *P6[3];  // persistent UNet features (q,k,v)
  for (int u = 0; u < 3; u++) {
    P3[u] = (float*)wsAlloc(F64 * 4);
    P4[u] = (float*)wsAlloc(F128 * 4);
    P6[u] = (float*)wsAlloc(F256 * 4);
  }
  float* O0 = (float*)wsAlloc(F64 * 4);
  float* O1 = (float*)wsAlloc(F128 * 4);
  float* O2 = (float*)wsAlloc(F256 * 4);
  float *S256[3], *S128[3], *S64p[3];
  for (int i = 0; i < 3; i++) S256[i] = (float*)wsAlloc(F256 * 4);
  for (int i = 0; i < 3; i++) S128[i] = (float*)wsAlloc(F128 * 4);
  for (int i = 0; i < 3; i++) S64p[i] = (float*)wsAlloc(F64 * 4);
  float* Shid = (float*)wsAlloc(2 * F256 * 4);           // MLP hidden (128ch)
  bf16_t* Gq = (bf16_t*)wsAlloc((size_t)16 * 512 * 1024 * 2);
  bf16_t* Gk = (bf16_t*)wsAlloc((size_t)16 * 512 * 1024 * 2);
  bf16_t* Gv = (bf16_t*)wsAlloc((size_t)16 * 512 * 1024 * 2);
  bf16_t* Gt = (bf16_t*)wsAlloc((size_t)16 * 512 * 512 * 2);
  float* part = (float*)wsAlloc(512 * 4);
  float* red = (float*)wsAlloc(4 * 4);

  // ---- launch helpers ----
  auto launch_conv = [&](const float* in, CpT p, const float* res, float* out,
                         int cin, int stride, bool up, int hin, int win, int cout,
                         bool relu) {
    int MT = (cout + 15) / 16;
    int KC = (cin * 9 + 31) / 32;
    int tot = KC * MT * 512;
    bf16_t* wp = (bf16_t*)wsAlloc((size_t)tot * 2);
    k_pack<<<(tot + 255) / 256, 256, 0, stream>>>(p.w, wp, cout, cin, 9, KC, MT);
    int hout = up ? hin * 2 : hin / stride;
    int wout = up ? win * 2 : win / stride;
    dim3 grid(wout / 64, hout, 2);
    int rl = relu ? 1 : 0;
    if (cin == 3)
      k_conv3x3<3, 1, false><<<grid, 128, 0, stream>>>(in, wp, p.b, res, out, hin,
                                                       win, hout, wout, cout, MT, rl);
    else if (up)
      k_conv3x3<64, 1, true><<<grid, 128, 0, stream>>>(in, wp, p.b, res, out, hin,
                                                       win, hout, wout, cout, MT, rl);
    else if (stride == 2)
      k_conv3x3<64, 2, false><<<grid, 128, 0, stream>>>(in, wp, p.b, res, out, hin,
                                                        win, hout, wout, cout, MT, rl);
    else
      k_conv3x3<64, 1, false><<<grid, 128, 0, stream>>>(in, wp, p.b, res, out, hin,
                                                        win, hout, wout, cout, MT, rl);
  };

  auto launch_pw = [&](const float* in, const float* w, const float* b,
                       const float* res, float* out, int cin, int cout, int HW,
                       bool relu) {
    int MT = (cout + 15) / 16;
    int KC = cin / 32;
    int tot = KC * MT * 512;
    bf16_t* wp = (bf16_t*)wsAlloc((size_t)tot * 2);
    k_pack<<<(tot + 255) / 256, 256, 0, stream>>>(w, wp, cout, cin, 1, KC, MT);
    dim3 grid(HW / 64, 1, 2);
    int rl = relu ? 1 : 0;
    if (cin == 64)
      k_pw<64><<<grid, 128, 0, stream>>>(in, wp, b, res, out, HW, cout, MT, rl);
    else
      k_pw<128><<<grid, 128, 0, stream>>>(in, wp, b, res, out, HW, cout, MT, rl);
  };

  // ---- UNet ----
  auto run_unet = [&](UnetT& u, const float* x, float* F3, float* F4, float* F6) {
    launch_conv(x, u.cin, nullptr, S256[0], 3, 1, false, 256, 256, 64, true);        // f0
    launch_conv(S256[0], u.c[0][0], nullptr, S128[0], 64, 2, false, 256, 256, 64, true);
    launch_conv(S128[0], u.c[0][1], nullptr, S128[1], 64, 1, false, 128, 128, 64, true);  // f1
    launch_conv(S128[1], u.c[1][0], nullptr, S64p[0], 64, 2, false, 128, 128, 64, true);
    launch_conv(S64p[0], u.c[1][1], nullptr, S64p[1], 64, 1, false, 64, 64, 64, true);    // f2
    launch_conv(S64p[1], u.c[2][0], nullptr, S64p[0], 64, 1, false, 64, 64, 64, true);
    launch_conv(S64p[0], u.c[2][1], S64p[1], F3, 64, 1, false, 64, 64, 64, true);         // f3
    launch_conv(F3, u.c[3][0], nullptr, S128[0], 64, 1, true, 64, 64, 64, true);          // up
    launch_conv(S128[0], u.c[3][1], S128[1], F4, 64, 1, false, 128, 128, 64, true);       // f4
    launch_conv(F4, u.c[4][0], nullptr, S256[1], 64, 1, true, 128, 128, 64, true);        // up
    launch_conv(S256[1], u.c[4][1], S256[0], S256[2], 64, 1, false, 256, 256, 64, true);  // f5
    launch_conv(S256[2], u.c[5][0], nullptr, S256[0], 64, 1, false, 256, 256, 64, true);
    launch_conv(S256[0], u.c[5][1], nullptr, F6, 64, 1, false, 256, 256, 64, true);       // f6
  };

  run_unet(uq, in_q, P3[0], P4[0], P6[0]);
  run_unet(uk, in_k, P3[1], P4[1], P6[1]);
  run_unet(uk, in_v, P3[2], P4[2], P6[2]);  // v uses unet_k params (per reference)

  // ---- Transformer unit (linear patch attention + MLP + GroupNorm) ----
  auto run_trans = [&](int hs, TransT& tp, const float* qf, const float* kf,
                       const float* vf, float* outb, float** T) {
    const int Sf = hs * hs;
    const int Sp = Sf / 64;  // patched sequence length
    const size_t tot = (size_t)2 * 64 * Sf;
    const size_t gt = (size_t)16 * 512 * Sp;
    // Q path
    k_addpe<<<(tot + 255) / 256, 256, 0, stream>>>(qf, T[0], hs, hs);
    launch_pw(T[0], tp.wq, nullptr, nullptr, T[1], 64, 64, Sf, false);
    k_gather<<<(gt + 255) / 256, 256, 0, stream>>>(T[1], Gq, hs, hs, 1);
    // K path
    k_addpe<<<(tot + 255) / 256, 256, 0, stream>>>(kf, T[0], hs, hs);
    launch_pw(T[0], tp.wk, nullptr, nullptr, T[1], 64, 64, Sf, false);
    k_gather<<<(gt + 255) / 256, 256, 0, stream>>>(T[1], Gk, hs, hs, 0);
    // V path (no PE)
    launch_pw(vf, tp.wv, nullptr, nullptr, T[0], 64, 64, Sf, false);
    k_gather<<<(gt + 255) / 256, 256, 0, stream>>>(T[0], Gv, hs, hs, 0);
    // tmp^T = (K @ V^T / S)^T  -> bf16, fused scale + transpose store
    k_gemm_nt<<<dim3(8, 8, 16), 128, 0, stream>>>(Gk, Gv, nullptr, Gt, 512, 512,
                                                  Sp, 1.f / (float)Sp);
    // out = tmp^T @ Q  (NT with Q stored transposed [s][c]) -> fp32 [bn][512][Sp]
    k_gemm_nt<<<dim3(Sp / 64, 8, 16), 128, 0, stream>>>(Gt, Gq, T[1], nullptr, 512,
                                                        Sp, 512, 1.f);
    k_scatter<<<(gt + 255) / 256, 256, 0, stream>>>(T[1], T[0], hs, hs);
    launch_pw(T[0], tp.fc, nullptr, nullptr, T[2], 64, 64, Sf, false);       // mha out
    launch_pw(T[2], tp.mlp0.w, tp.mlp0.b, nullptr, Shid, 64, 128, Sf, true);
    launch_pw(Shid, tp.mlp1.w, tp.mlp1.b, T[2], T[0], 128, 64, Sf, false);   // + mha
    // GroupNorm(1)
    int n = 64 * Sf;
    k_gn_part<<<dim3(128, 2), 256, 0, stream>>>(T[0], part, n);
    k_gn_fin<<<1, 32, 0, stream>>>(part, red);
    k_gn_apply<<<(tot + 255) / 256, 256, 0, stream>>>(T[0], red, tp.gn_w, tp.gn_b,
                                                      outb, n, Sf);
  };

  run_trans(64, tr[0], P3[0], P3[1], P3[2], O0, S64p);
  run_trans(128, tr[1], P4[0], P4[1], P4[2], O1, S128);
  run_trans(256, tr[2], P6[0], P6[1], P6[2], O2, S256);

  // ---- decoder ----
  launch_conv(O2, convd[0][0], nullptr, S256[0], 64, 1, false, 256, 256, 64, true);
  launch_conv(S256[0], convd[0][1], nullptr, S256[1], 64, 1, false, 256, 256, 64, true);
  launch_conv(S256[1], convd[1][0], nullptr, S128[0], 64, 2, false, 256, 256, 64, true);
  launch_conv(S128[0], convd[1][1], O1, S128[1], 64, 1, false, 128, 128, 64, true);
  launch_conv(S128[1], convd[2][0], nullptr, S64p[0], 64, 2, false, 128, 128, 64, true);
  launch_conv(S64p[0], convd[2][1], O0, S64p[1], 64, 1, false, 64, 64, 64, true);
  launch_conv(S64p[1], convd[3][0], nullptr, S64p[0], 64, 1, false, 64, 64, 64, true);
  launch_conv(S64p[0], convd[3][1], S64p[1], S64p[2], 64, 1, false, 64, 64, 64, true);
  launch_conv(S64p[2], convd[4][0], nullptr, S128[0], 64, 1, true, 64, 64, 64, true);
  launch_conv(S128[0], convd[4][1], nullptr, S128[1], 64, 1, false, 128, 128, 64, true);
  launch_conv(S128[1], convd[5][0], nullptr, S256[0], 64, 1, true, 128, 128, 64, true);
  launch_conv(S256[0], convd[5][1], nullptr, S256[1], 64, 1, false, 256, 256, 64, true);
  launch_conv(S256[1], conv6p, nullptr, S256[0], 64, 1, false, 256, 256, 64, true);
  launch_conv(S256[0], convOutp, nullptr, (float*)d_out, 64, 1, false, 256, 256, 3,
              false);
}
